// Attention_197568495628
// MI455X (gfx1250) — matrix operand order repaired and checked by reading, not compiled, not run on hardware
//
#include <hip/hip_runtime.h>
#include <hip/hip_bf16.h>
#include <math.h>

typedef _Float16 v16h __attribute__((ext_vector_type(16)));
typedef _Float16 v8h  __attribute__((ext_vector_type(8)));
typedef float    v8f  __attribute__((ext_vector_type(8)));
typedef float    v4f  __attribute__((ext_vector_type(4)));

#define U_DIM 512
#define N_TOK 2048
#define B_SZ  8
#define M_TOT (B_SZ * N_TOK)   // 16384 token rows

static __device__ inline v8f zero8() {
    v8f z;
#pragma unroll
    for (int i = 0; i < 8; ++i) z[i] = 0.0f;
    return z;
}

static __device__ inline v8f wmma_f16(v16h a, v16h b, v8f c) {
    // D = A(16x32 f16) * B(32x16 f16) + C(16x16 f32)
    return __builtin_amdgcn_wmma_f32_16x16x32_f16(
        /*neg_a=*/false, a, /*neg_b=*/false, b,
        /*c_mod=*/(short)0, c, /*reuse_a=*/false, /*reuse_b=*/false);
}

// A-fragment (16x32, MxK) from a row-major f16 matrix (ld elems/row).
// Lane layout: row = lane&15; lanes 0-15 hold K = k0+{0..7, 16..23},
// lanes 16-31 hold K = k0+8+{0..7, 16..23}.
static __device__ inline v16h load_a_f16(const _Float16* base, int ld, int row0, int k0) {
    int lane = threadIdx.x & 31;
    int m  = row0 + (lane & 15);
    int kb = k0 + ((lane & 16) ? 8 : 0);
    const _Float16* p = base + (size_t)m * ld + kb;
    v8h lo = *(const v8h*)p;          // K = kb .. kb+7
    v8h hi = *(const v8h*)(p + 16);   // K = kb+16 .. kb+23
    v16h r;
#pragma unroll
    for (int i = 0; i < 8; ++i) { r[i] = lo[i]; r[i + 8] = hi[i]; }
    return r;
}

// B-fragment (32x16, KxN) from a K-contiguous layout: base[n*ld + k].
// Lane layout: col = lane&15; lanes 0-15 hold K = k0+{0..7, 16..23},
// lanes 16-31 hold K = k0+8+{0..7, 16..23}.
static __device__ inline v16h load_b_kc(const _Float16* base, int ld, int col0, int k0) {
    int lane = threadIdx.x & 31;
    int n  = col0 + (lane & 15);
    int kb = k0 + ((lane & 16) ? 8 : 0);
    const _Float16* p = base + (size_t)n * ld + kb;
    v8h lo = *(const v8h*)p;          // K = kb .. kb+7
    v8h hi = *(const v8h*)(p + 16);   // K = kb+16 .. kb+23
    v16h r;
#pragma unroll
    for (int i = 0; i < 8; ++i) { r[i] = lo[i]; r[i + 8] = hi[i]; }
    return r;
}

// ---------------------------------------------------------------------------
// Kernel 0a: transpose + convert the four weight matrices to f16, K-contiguous:
// WT[n][k] = (f16) W[k][n]
// ---------------------------------------------------------------------------
__global__ __launch_bounds__(256) void k_prep(
    const float* __restrict__ Wq, const float* __restrict__ Wk,
    const float* __restrict__ Wv, const float* __restrict__ Wo,
    _Float16* __restrict__ WqT, _Float16* __restrict__ WkT,
    _Float16* __restrict__ WvT, _Float16* __restrict__ WoT)
{
    int idx = blockIdx.x * 256 + threadIdx.x;   // 0 .. 512*512-1
    int kk = idx >> 9;
    int n  = idx & 511;
    int src = kk * U_DIM + n;
    int dst = n * U_DIM + kk;
    WqT[dst] = (_Float16)Wq[src];
    WkT[dst] = (_Float16)Wk[src];
    WvT[dst] = (_Float16)Wv[src];
    WoT[dst] = (_Float16)Wo[src];
}

// ---------------------------------------------------------------------------
// Kernel 0b: convert x (fp32) -> xh (f16) once. 8 elements per thread.
// ---------------------------------------------------------------------------
__global__ __launch_bounds__(256) void k_xconv(
    const float* __restrict__ x, _Float16* __restrict__ xh)
{
    size_t base = ((size_t)blockIdx.x * 256 + threadIdx.x) * 8;
    v4f a = *(const v4f*)(x + base);
    v4f b = *(const v4f*)(x + base + 4);
    v8h o;
#pragma unroll
    for (int i = 0; i < 4; ++i) { o[i] = (_Float16)a[i]; o[4 + i] = (_Float16)b[i]; }
    *(v8h*)(xh + base) = o;
}

// ---------------------------------------------------------------------------
// Kernel 1: fused QKV projection. One wave computes the SAME 16x16 tile of
// q, k and v (3 accumulators share one A-fragment -> 3x less A traffic).
// q,k stored row-major f16 [M_TOT][U]; v stored transposed vT[b][u][token]
// so the P@v B-operand is K-contiguous.
// ---------------------------------------------------------------------------
__global__ __launch_bounds__(128) void k_qkv(
    const _Float16* __restrict__ xh,
    const _Float16* __restrict__ WqT, const _Float16* __restrict__ WkT,
    const _Float16* __restrict__ WvT,
    const float* __restrict__ bq, const float* __restrict__ bk,
    const float* __restrict__ bv,
    _Float16* __restrict__ q, _Float16* __restrict__ k,
    _Float16* __restrict__ vT)
{
    int wid = blockIdx.x * 4 + (threadIdx.x >> 5);
    int rt = wid / (U_DIM / 16);
    int ct = wid % (U_DIM / 16);
    int row0 = rt * 16, col0 = ct * 16;

    v8f aq = zero8(), ak = zero8(), av = zero8();
#pragma unroll
    for (int ku = 0; ku < U_DIM; ku += 32) {
        v16h a = load_a_f16(xh, U_DIM, row0, ku);
        aq = wmma_f16(a, load_b_kc(WqT, U_DIM, col0, ku), aq);
        ak = wmma_f16(a, load_b_kc(WkT, U_DIM, col0, ku), ak);
        av = wmma_f16(a, load_b_kc(WvT, U_DIM, col0, ku), av);
    }

    int lane = threadIdx.x & 31;
    int col  = col0 + (lane & 15);
    int roff = (lane & 16) ? 8 : 0;
    float bq_ = bq[col], bk_ = bk[col], bv_ = bv[col];

    // q, k: row-major scalar stores
#pragma unroll
    for (int i = 0; i < 8; ++i) {
        size_t idx = (size_t)(row0 + roff + i) * U_DIM + col;
        q[idx] = (_Float16)(aq[i] + bq_);
        k[idx] = (_Float16)(ak[i] + bk_);
    }
    // v: transposed packed store: (token m, feature col) -> vT[(b*U+col)*N + m%N]
    int grow = row0 + roff;                // token row of av[0] (8 consecutive)
    int bidx = grow / N_TOK;
    int nloc = grow % N_TOK;
    v8h pk;
#pragma unroll
    for (int i = 0; i < 8; ++i) pk[i] = (_Float16)(av[i] + bv_);
    *(v8h*)(vT + ((size_t)bidx * U_DIM + col) * N_TOK + nloc) = pk;
}

// ---------------------------------------------------------------------------
// Kernel 2: attention core. One wave per (batch, 16-query block).
//   phase 1: S[16][2048] = (q @ k^T) * scale  -> LDS (f16), k prefetched ahead
//   phase 2: in-place row softmax: 32 lanes (half-row each), b128 LDS access,
//            cross-half combine via wave32 __shfl_xor(...,16)
//   phase 3: ctx[16][512] = P @ v  (A-fragments straight from LDS, v prefetch)
// ---------------------------------------------------------------------------
__global__ __launch_bounds__(32) void k_attn(
    const _Float16* __restrict__ q, const _Float16* __restrict__ kmat,
    const _Float16* __restrict__ vT, _Float16* __restrict__ ctx)
{
    __shared__ __align__(16) _Float16 S[16 * N_TOK];   // 64 KB

    int wid  = blockIdx.x;                 // 0 .. B*N/16-1
    int bidx = wid / (N_TOK / 16);
    int qt   = wid % (N_TOK / 16);
    int lane = threadIdx.x & 31;
    int qrow0 = bidx * N_TOK + qt * 16;    // global token row of this q block

    // Preload the 16 q A-fragments for the whole row-block (128 VGPRs).
    v16h qf[16];
#pragma unroll
    for (int t = 0; t < 16; ++t) qf[t] = load_a_f16(q, U_DIM, qrow0, t * 32);

    const float scale = 0.04419417382415922f;   // 1/sqrt(512)
    const _Float16* kb_base = kmat + (size_t)bidx * N_TOK * U_DIM;

    // ---- phase 1: scores ----
    for (int j = 0; j < N_TOK / 16; ++j) {
        // prefetch next key tile's rows (L2 -> L0) while this tile computes
        if (j + 1 < N_TOK / 16) {
            const _Float16* pf = kb_base
                + (size_t)((j + 1) * 16 + (lane & 15)) * U_DIM
                + ((lane & 16) ? 256 : 0);
            __builtin_prefetch(pf, 0, 3);
        }
        v8f acc = zero8();
#pragma unroll
        for (int t = 0; t < 16; ++t) {
            v16h b = load_b_kc(kb_base, U_DIM, j * 16, t * 32);
            acc = wmma_f16(qf[t], b, acc);
        }
        int col  = j * 16 + (lane & 15);
        int roff = (lane & 16) ? 8 : 0;
#pragma unroll
        for (int i = 0; i < 8; ++i)
            S[(roff + i) * N_TOK + col] = (_Float16)(acc[i] * scale);
    }
    __syncthreads();

    // ---- phase 2: row softmax, all 32 lanes active ----
    {
        int r = lane & 15;                       // row
        int cb = (lane & 16) ? (N_TOK / 2) : 0;  // column half
        _Float16* row = S + r * N_TOK + cb;

        float mx = -1e30f;
        for (int c = 0; c < N_TOK / 2; c += 8) {
            v8h v = *(const v8h*)(row + c);
#pragma unroll
            for (int i = 0; i < 8; ++i) mx = fmaxf(mx, (float)v[i]);
        }
        mx = fmaxf(mx, __shfl_xor(mx, 16, 32));

        float sum = 0.0f;
        for (int c = 0; c < N_TOK / 2; c += 8) {
            v8h v = *(const v8h*)(row + c);
            v8h e;
#pragma unroll
            for (int i = 0; i < 8; ++i) {
                float ev = __expf((float)v[i] - mx);
                sum += ev;
                e[i] = (_Float16)ev;
            }
            *(v8h*)(row + c) = e;
        }
        sum += __shfl_xor(sum, 16, 32);
        float inv = 1.0f / sum;

        for (int c = 0; c < N_TOK / 2; c += 8) {
            v8h v = *(const v8h*)(row + c);
#pragma unroll
            for (int i = 0; i < 8; ++i) v[i] = (_Float16)((float)v[i] * inv);
            *(v8h*)(row + c) = v;
        }
    }
    __syncthreads();

    // ---- phase 3: ctx = P @ v ----
    const _Float16* vb = vT + (size_t)bidx * U_DIM * N_TOK;
    for (int c = 0; c < U_DIM / 16; ++c) {
        v8f acc = zero8();
        for (int kb2 = 0; kb2 < N_TOK / 32; ++kb2) {
            // prefetch next v tile for this column strip
            if (kb2 + 1 < N_TOK / 32) {
                const _Float16* pf = vb
                    + (size_t)(c * 16 + (lane & 15)) * N_TOK
                    + (kb2 + 1) * 32 + ((lane & 16) ? 16 : 0);
                __builtin_prefetch(pf, 0, 3);
            }
            // A-fragment from LDS (P is row-major [16][2048] f16)
            int m    = lane & 15;
            int koff = kb2 * 32 + ((lane & 16) ? 8 : 0);
            const _Float16* p = S + m * N_TOK + koff;
            v8h lo = *(const v8h*)p;
            v8h hi = *(const v8h*)(p + 16);
            v16h a;
#pragma unroll
            for (int i = 0; i < 8; ++i) { a[i] = lo[i]; a[i + 8] = hi[i]; }
            v16h b = load_b_kc(vb, N_TOK, c * 16, kb2 * 32);
            acc = wmma_f16(a, b, acc);
        }
        int col  = c * 16 + (lane & 15);
        int roff = (lane & 16) ? 8 : 0;
#pragma unroll
        for (int i = 0; i < 8; ++i)
            ctx[(size_t)(qrow0 + roff + i) * U_DIM + col] = (_Float16)acc[i];
    }
}

// ---------------------------------------------------------------------------
// Kernel 3: out = x + ctx @ Wo + bo   (fp32 output)
// ---------------------------------------------------------------------------
__global__ __launch_bounds__(128) void k_out(
    const float* __restrict__ x, const _Float16* __restrict__ ctx,
    const _Float16* __restrict__ WoT, const float* __restrict__ bo,
    float* __restrict__ out)
{
    int wid = blockIdx.x * 4 + (threadIdx.x >> 5);
    int rt = wid / (U_DIM / 16);
    int ct = wid % (U_DIM / 16);
    int row0 = rt * 16, col0 = ct * 16;

    v8f acc = zero8();
#pragma unroll
    for (int ku = 0; ku < U_DIM; ku += 32) {
        v16h a = load_a_f16(ctx, U_DIM, row0, ku);
        v16h b = load_b_kc(WoT, U_DIM, col0, ku);
        acc = wmma_f16(a, b, acc);
    }

    int lane = threadIdx.x & 31;
    int col  = col0 + (lane & 15);
    int roff = (lane & 16) ? 8 : 0;
    float bval = bo[col];
#pragma unroll
    for (int i = 0; i < 8; ++i) {
        size_t idx = (size_t)(row0 + roff + i) * U_DIM + col;
        out[idx] = x[idx] + acc[i] + bval;
    }
}

// ---------------------------------------------------------------------------
extern "C" void kernel_launch(void* const* d_in, const int* in_sizes, int n_in,
                              void* d_out, int out_size, void* d_ws, size_t ws_size,
                              hipStream_t stream) {
    const float* x  = (const float*)d_in[0];
    const float* Wq = (const float*)d_in[1];
    const float* bq = (const float*)d_in[2];
    const float* Wk = (const float*)d_in[3];
    const float* bk = (const float*)d_in[4];
    const float* Wv = (const float*)d_in[5];
    const float* bv = (const float*)d_in[6];
    const float* Wo = (const float*)d_in[7];
    const float* bo = (const float*)d_in[8];
    float* out = (float*)d_out;

    // Workspace layout (f16): 4 weight transposes (2 MB) + xh,q,k,vT,ctx (16 MB each)
    char* ws = (char*)d_ws;
    const size_t WSZ = (size_t)U_DIM * U_DIM;        // elems per weight matrix
    const size_t ASZ = (size_t)M_TOT * U_DIM;        // elems per activation
    _Float16* WqT = (_Float16*)ws;
    _Float16* WkT = WqT + WSZ;
    _Float16* WvT = WkT + WSZ;
    _Float16* WoT = WvT + WSZ;
    _Float16* xh  = WoT + WSZ;
    _Float16* qf  = xh + ASZ;
    _Float16* kf  = qf + ASZ;
    _Float16* vT  = kf + ASZ;
    _Float16* ctx = vT + ASZ;

    // K0a: weight transpose/convert
    k_prep<<<(U_DIM * U_DIM) / 256, 256, 0, stream>>>(Wq, Wk, Wv, Wo,
                                                      WqT, WkT, WvT, WoT);
    // K0b: x -> f16 (8 elems per thread)
    k_xconv<<<(int)(ASZ / (256 * 8)), 256, 0, stream>>>(x, xh);
    // K1: fused QKV projection. 1024*32 tiles, 4 waves per block.
    k_qkv<<<((M_TOT / 16) * (U_DIM / 16)) / 4, 128, 0, stream>>>(
        xh, WqT, WkT, WvT, bq, bk, bv, qf, kf, vT);
    // K2: attention core. One wave per (batch, 16-query block).
    k_attn<<<B_SZ * (N_TOK / 16), 32, 0, stream>>>(qf, kf, vT, ctx);
    // K3: output projection + residual. 1024*32 tiles, 4 waves per block.
    k_out<<<((M_TOT / 16) * (U_DIM / 16)) / 4, 128, 0, stream>>>(
        x, ctx, WoT, bo, out);
}
